// RGCNModel_72292889526583
// MI455X (gfx1250) — compile-verified, matrix operand order checked by reading, not run on hardware
//
#include <hip/hip_runtime.h>

#define NENT  40000
#define NRELS 20
#define DIM   128

typedef __attribute__((ext_vector_type(16))) _Float16 v16h;
typedef __attribute__((ext_vector_type(8)))  _Float16 v8h;
typedef __attribute__((ext_vector_type(8)))  float    v8f;

// ---------------------------------------------------------------------------
// WMMA fragment load for 16-bit A/B operands of V_WMMA_F32_16X16X32_F16.
// Per CDNA5 ISA §7.12.2 the 16 halves a lane holds from its
// row-major 32-half K-slice are two contiguous 8-half runs:
//   frag[0..7]  = row[ko .. ko+7]          (VGPRs 0-3: K = ko..ko+7)
//   frag[8..15] = row[16+ko .. 16+ko+7]    (VGPRs 4-7: K = 16+ko..16+ko+7)
// with ko = (lane>=16) ? 8 : 0.  => exactly two global_load_b128 per fragment.
// ---------------------------------------------------------------------------
__device__ __forceinline__ v16h load_frag(const _Float16* __restrict__ rowp, int lane) {
  const int ko = (lane & 16) ? 8 : 0;
  union { v16h v; v8h h[2]; } u;
  u.h[0] = *(const v8h*)(rowp + ko);
  u.h[1] = *(const v8h*)(rowp + 16 + ko);
  return u.v;
}

// ---------------------------------------------------------------------------
// Edge preprocessing: counting-sort edges by relation (padded to multiples of
// 16 with -1 sentinels) + float in-degree. Shared by both layers.
// ---------------------------------------------------------------------------
__global__ void k_init(int* __restrict__ perm, int pmax, float* __restrict__ deg,
                       int* __restrict__ cnt, int* __restrict__ cursor) {
  const int idx = blockIdx.x * blockDim.x + threadIdx.x;
  if (idx < pmax)  perm[idx] = -1;
  if (idx < NENT)  deg[idx] = 0.0f;
  if (idx < NRELS) { cnt[idx] = 0; cursor[idx] = 0; }
}

__global__ void k_count_deg(const int* __restrict__ et, const int* __restrict__ dst,
                            int E, int* __restrict__ cnt, float* __restrict__ deg) {
  const int e = blockIdx.x * blockDim.x + threadIdx.x;
  if (e >= E) return;
  atomicAdd(&cnt[et[e]], 1);
  atomicAdd(&deg[dst[e]], 1.0f);
}

__global__ void k_offsets(const int* __restrict__ cnt, int* __restrict__ offr) {
  if (blockIdx.x == 0 && threadIdx.x == 0) {
    int p = 0;
    for (int r = 0; r < NRELS; ++r) {
      offr[r] = p;
      p += (cnt[r] + 15) & ~15;   // pad each relation bucket to 16
    }
  }
}

__global__ void k_place(const int* __restrict__ et, int E, const int* __restrict__ offr,
                        int* __restrict__ cursor, int* __restrict__ perm) {
  const int e = blockIdx.x * blockDim.x + threadIdx.x;
  if (e >= E) return;
  const int r = et[e];
  const int pos = offr[r] + atomicAdd(&cursor[r], 1);
  perm[pos] = e;
}

// ---------------------------------------------------------------------------
// Weight prep: W_r = sum_b comp[r,b] * basis[b]  stored transposed+f16 as
// Wt[r][out][in] so B fragments load contiguously. Same for rootT.
// ---------------------------------------------------------------------------
__global__ void k_weights(const float* __restrict__ comp, const float* __restrict__ basis,
                          int Bb, _Float16* __restrict__ Wt) {
  const int idx = blockIdx.x * blockDim.x + threadIdx.x;
  if (idx >= NRELS * DIM * DIM) return;
  const int r   = idx / (DIM * DIM);
  const int rem = idx - r * DIM * DIM;
  const int o   = rem >> 7;     // out index
  const int ii  = rem & 127;    // in index
  float s = 0.0f;
  for (int b = 0; b < Bb; ++b)
    s += comp[r * Bb + b] * basis[((size_t)b * DIM + ii) * DIM + o];
  Wt[idx] = (_Float16)s;        // idx == r*16384 + o*128 + ii
}

__global__ void k_castroot(const float* __restrict__ root, _Float16* __restrict__ rootT) {
  const int idx = blockIdx.x * blockDim.x + threadIdx.x;
  if (idx >= DIM * DIM) return;
  const int o = idx >> 7, ii = idx & 127;
  rootT[idx] = (_Float16)root[ii * DIM + o];
}

__global__ void k_castx(const float* __restrict__ x, _Float16* __restrict__ xh, int n) {
  const int idx = blockIdx.x * blockDim.x + threadIdx.x;
  if (idx < n) xh[idx] = (_Float16)x[idx];
}

__global__ void k_zero_f32(float* __restrict__ p, int n) {
  const int idx = blockIdx.x * blockDim.x + threadIdx.x;
  if (idx < n) p[idx] = 0.0f;
}

// ---------------------------------------------------------------------------
// Core kernel: per-wave group of 16 same-relation edges.
//   A = xh[src rows] (16x128 f16, preloaded in 4 fragments),
//   B = Wt[r] (128x128 pre-transposed), processed nt-outer so only one v8f
//   accumulator is live: 4 chained WMMAs per 16-wide column tile, then that
//   tile's f32 atomic scatter into agg[dst]. Sentinel (-1) slots contribute
//   exact zeros (their A rows are zeroed), so the scatter is BRANCHLESS:
//   their atomics are redirected to a trash row agg[NENT] via v_cndmask,
//   avoiding 64 exec-mask branch sequences per group.
// ---------------------------------------------------------------------------
__global__ void __launch_bounds__(256)
k_edge_gemm(const int* __restrict__ perm, int ngroups,
            const int* __restrict__ srcA, const int* __restrict__ dstA,
            const int* __restrict__ etA,
            const _Float16* __restrict__ xh,
            const _Float16* __restrict__ Wt,
            float* __restrict__ agg) {
  const int lane = threadIdx.x & 31;
  const int g = blockIdx.x * (blockDim.x >> 5) + (threadIdx.x >> 5);
  if (g >= ngroups) return;                       // uniform per wave
  const int i = lane & 15;                        // edge slot / N column
  const int e = perm[g * 16 + i];
  const bool valid = (e >= 0);
  int srcn = 0, dstn = -1, r = 0;
  if (valid) { srcn = srcA[e]; dstn = dstA[e]; r = etA[e]; }

  const unsigned mlo = (unsigned)__ballot(valid);
  if (mlo == 0u) return;                          // fully-padded tail group (uniform)
  const int fl = __ffs(mlo) - 1;
  const int ru = __shfl(r, fl, 32);               // uniform relation of this group
  const _Float16* __restrict__ Wr = Wt + (size_t)ru * (DIM * DIM);
  const _Float16* __restrict__ arow = xh + (size_t)srcn * DIM;

  // Preload all four A fragments (K = 0..127), reused by every column tile.
  v16h a4[4];
#pragma unroll
  for (int ks = 0; ks < 4; ++ks) {
    a4[ks] = {};
    if (valid) a4[ks] = load_frag(arow + ks * 32, lane);
  }

  // Destination rows for the C/D layout: result row m lives in acc[j] with
  // m = j + 8*(lane>=16); lanes 0..15 carry edges 0..15's dst.
  // Sentinel rows -> trash row NENT (their acc values are exactly 0.0f).
  int dmv[8];
  const int mbase = (lane & 16) ? 8 : 0;
#pragma unroll
  for (int j = 0; j < 8; ++j) {
    const int dm = __shfl(dstn, j + mbase, 32);
    dmv[j] = (dm < 0) ? NENT : dm;
  }

#pragma unroll
  for (int nt = 0; nt < 8; ++nt) {                // 8 output column tiles of 16
    const _Float16* __restrict__ brow = Wr + (size_t)(nt * 16 + i) * DIM;
    v8f acc = {};
#pragma unroll
    for (int ks = 0; ks < 4; ++ks) {              // K = 128 in 4 steps of 32
      const v16h b = load_frag(brow + ks * 32, lane);
      acc = __builtin_amdgcn_wmma_f32_16x16x32_f16(
          false, a4[ks], false, b, (short)0, acc, false, false);
    }
    float* __restrict__ colp = agg + (size_t)(nt * 16 + i);
#pragma unroll
    for (int j = 0; j < 8; ++j)
      atomicAdd(colp + (size_t)dmv[j] * DIM, acc[j]);   // branchless
  }
}

// ---------------------------------------------------------------------------
// Finalize: out = act(agg/deg + x@root + bias). Root GEMM is a 16-row WMMA
// tile per wave; writes f32 result (layer 2 -> d_out) and/or f16 copy for the
// next layer's A operand.
// ---------------------------------------------------------------------------
__global__ void __launch_bounds__(256)
k_finalize(const _Float16* __restrict__ xh_in,
           const _Float16* __restrict__ rootT,
           const float* __restrict__ agg,
           const float* __restrict__ deg,
           const float* __restrict__ bias,
           float* __restrict__ out_f32,
           _Float16* __restrict__ xh_out,
           int do_tanh, int ngroups) {
  const int lane = threadIdx.x & 31;
  const int g = blockIdx.x * (blockDim.x >> 5) + (threadIdx.x >> 5);
  if (g >= ngroups) return;
  const int i = lane & 15;
  const int base = g * 16;

  v8f acc[8] = {};
#pragma unroll
  for (int ks = 0; ks < 4; ++ks) {
    const v16h a = load_frag(xh_in + (size_t)(base + i) * DIM + ks * 32, lane);
#pragma unroll
    for (int nt = 0; nt < 8; ++nt) {
      const v16h b = load_frag(rootT + (size_t)(nt * 16 + i) * DIM + ks * 32, lane);
      acc[nt] = __builtin_amdgcn_wmma_f32_16x16x32_f16(
          false, a, false, b, (short)0, acc[nt], false, false);
    }
  }

#pragma unroll
  for (int j = 0; j < 8; ++j) {
    const int n = base + j + ((lane & 16) ? 8 : 0);
    float dg = deg[n];
    dg = (dg < 1.0f) ? 1.0f : dg;                 // clip(deg, 1)
    const float inv = 1.0f / dg;
#pragma unroll
    for (int nt = 0; nt < 8; ++nt) {
      const int col = nt * 16 + i;
      float v = acc[nt][j] + agg[(size_t)n * DIM + col] * inv + bias[col];
      if (do_tanh) v = tanhf(v);
      if (out_f32) out_f32[(size_t)n * DIM + col] = v;
      if (xh_out)  xh_out[(size_t)n * DIM + col] = (_Float16)v;
    }
  }
}

// Tiny relation transform: r = init_rel @ w_rel  (40x128x128, negligible)
__global__ void k_rel(const float* __restrict__ a, const float* __restrict__ w,
                      float* __restrict__ out, int rows) {
  const int idx = blockIdx.x * blockDim.x + threadIdx.x;
  if (idx >= rows * DIM) return;
  const int i = idx >> 7, o = idx & 127;
  float s = 0.0f;
  for (int k = 0; k < DIM; ++k) s += a[i * DIM + k] * w[k * DIM + o];
  out[idx] = s;
}

// ---------------------------------------------------------------------------
extern "C" void kernel_launch(void* const* d_in, const int* in_sizes, int n_in,
                              void* d_out, int out_size, void* d_ws, size_t ws_size,
                              hipStream_t stream) {
  (void)n_in; (void)out_size; (void)ws_size;

  const int E = in_sizes[1];
  const int* edge_index = (const int*)d_in[0];
  const int* src = edge_index;           // edge_index[0]
  const int* dst = edge_index + E;       // edge_index[1]
  const int* et  = (const int*)d_in[1];
  const float* x0       = (const float*)d_in[2];
  const float* init_rel = (const float*)d_in[3];
  const float* w_rel    = (const float*)d_in[4];
  const float* comp1  = (const float*)d_in[5];
  const float* basis1 = (const float*)d_in[6];
  const float* root1  = (const float*)d_in[7];
  const float* bias1  = (const float*)d_in[8];
  const float* comp2  = (const float*)d_in[9];
  const float* basis2 = (const float*)d_in[10];
  const float* root2  = (const float*)d_in[11];
  const float* bias2  = (const float*)d_in[12];
  const int Bb    = in_sizes[5] / NRELS;   // 8 basis matrices
  const int nrel2 = in_sizes[3] / DIM;     // 40 relation embeddings

  const int Pmax = E + 16 * NRELS;         // padded edge-perm capacity
  const int AGG_ROWS = NENT + 1;           // +1 trash row for sentinel scatter

  // Workspace carve (256B aligned): ~44 MB total
  char* w = (char*)d_ws;
  size_t off = 0;
  auto carve = [&](size_t bytes) -> char* {
    char* p = w + off;
    off += (bytes + 255) & ~(size_t)255;
    return p;
  };
  float*    agg    = (float*)   carve((size_t)AGG_ROWS * DIM * 4);
  _Float16* xh1    = (_Float16*)carve((size_t)NENT * DIM * 2);
  _Float16* xh2    = (_Float16*)carve((size_t)NENT * DIM * 2);
  _Float16* Wt     = (_Float16*)carve((size_t)NRELS * DIM * DIM * 2);
  _Float16* rootT  = (_Float16*)carve((size_t)DIM * DIM * 2);
  float*    deg    = (float*)   carve((size_t)NENT * 4);
  int*      perm   = (int*)     carve((size_t)Pmax * 4);
  int*      cnt    = (int*)     carve(NRELS * 4);
  int*      cursor = (int*)     carve(NRELS * 4);
  int*      offr   = (int*)     carve(NRELS * 4);

  float* out_x = (float*)d_out;
  float* out_r = out_x + (size_t)NENT * DIM;

  const int TB = 256;
  const int ngroups    = Pmax / 16;     // edge groups (incl. padded tails)
  const int fin_groups = NENT / 16;     // node row tiles
  const int aggN       = AGG_ROWS * DIM;

  // --- edge bucketing + degree (shared by both layers) ---
  k_init     <<<(Pmax + TB - 1) / TB, TB, 0, stream>>>(perm, Pmax, deg, cnt, cursor);
  k_count_deg<<<(E + TB - 1) / TB,   TB, 0, stream>>>(et, dst, E, cnt, deg);
  k_offsets  <<<1, 1, 0, stream>>>(cnt, offr);
  k_place    <<<(E + TB - 1) / TB,   TB, 0, stream>>>(et, E, offr, cursor, perm);

  // --- layer 1: init_embed -> xh2 (tanh) ---
  k_weights  <<<(NRELS * DIM * DIM + TB - 1) / TB, TB, 0, stream>>>(comp1, basis1, Bb, Wt);
  k_castroot <<<(DIM * DIM + TB - 1) / TB,         TB, 0, stream>>>(root1, rootT);
  k_castx    <<<(NENT * DIM + TB - 1) / TB,        TB, 0, stream>>>(x0, xh1, NENT * DIM);
  k_zero_f32 <<<(aggN + TB - 1) / TB,              TB, 0, stream>>>(agg, aggN);
  k_edge_gemm<<<(ngroups + 7) / 8,    TB, 0, stream>>>(perm, ngroups, src, dst, et, xh1, Wt, agg);
  k_finalize <<<(fin_groups + 7) / 8, TB, 0, stream>>>(xh1, rootT, agg, deg, bias1,
                                                       nullptr, xh2, 1, fin_groups);

  // --- layer 2: xh2 -> d_out (no act) ---
  k_weights  <<<(NRELS * DIM * DIM + TB - 1) / TB, TB, 0, stream>>>(comp2, basis2, Bb, Wt);
  k_castroot <<<(DIM * DIM + TB - 1) / TB,         TB, 0, stream>>>(root2, rootT);
  k_zero_f32 <<<(aggN + TB - 1) / TB,              TB, 0, stream>>>(agg, aggN);
  k_edge_gemm<<<(ngroups + 7) / 8,    TB, 0, stream>>>(perm, ngroups, src, dst, et, xh2, Wt, agg);
  k_finalize <<<(fin_groups + 7) / 8, TB, 0, stream>>>(xh2, rootT, agg, deg, bias2,
                                                       out_x, nullptr, 0, fin_groups);

  // --- relation embedding transform ---
  k_rel<<<(nrel2 * DIM + TB - 1) / TB, TB, 0, stream>>>(init_rel, w_rel, out_r, nrel2);
}
